// DynamicEdgeConvNet_5566277616468
// MI455X (gfx1250) — matrix operand order repaired and checked by reading, not compiled, not run on hardware
//
#include <hip/hip_runtime.h>
#include <float.h>
#include <math.h>

// Problem constants (match reference)
#define BGR 32
#define NPT 2048
#define KNN 20
#define HID 64
#define NCLS 40
#define BN_ (BGR*NPT)     // 65536 nodes
#define E1  (BN_*KNN)     // 1310720 edges

typedef __attribute__((ext_vector_type(16))) _Float16 v16h;
typedef __attribute__((ext_vector_type(8)))  float    v8f;
typedef _Float16 h16;

#define NEG_INF_FLIP 0x00800000u   // flipF(-FLT_MAX)

// ---------------------------------------------------------------- utilities
// order-preserving float<->uint transform (branch-free): float max == unsigned max
__device__ inline unsigned flipF(float f) {
    unsigned u = __float_as_uint(f);
    return u ^ (unsigned)(((int)u >> 31) | 0x80000000);
}
__device__ inline float unflipF(unsigned u) {
    return __uint_as_float(u ^ (unsigned)((~((int)u >> 31)) | 0x80000000));
}

// async global -> LDS copy of 16 bytes (CDNA5 path; bypasses VGPRs, ASYNCcnt-tracked)
__device__ inline void asyncCopy16(const void* gsrc, void* lds) {
    unsigned loff = (unsigned)(size_t)lds;   // generic LDS addr: low 32 bits = LDS offset
    asm volatile("global_load_async_to_lds_b128 %0, %1, off"
                 :: "v"(loff), "v"(gsrc) : "memory");
}
__device__ inline void asyncWait0() {
    asm volatile("s_wait_asynccnt 0x0" ::: "memory");
}

__global__ void k_fillu(unsigned* p, unsigned v, int n) {
    int i = blockIdx.x * blockDim.x + threadIdx.x;
    if (i < n) p[i] = v;
}

// --------------------- pre-swizzle weights into WMMA B-fragment layout (f16)
__global__ void k_wfrag(const float* __restrict__ w, h16* __restrict__ frag,
                        int ksteps, int ntiles, int ncols)
{
    int tid = blockIdx.x * blockDim.x + threadIdx.x;
    int total = ksteps * ntiles * 512;
    if (tid >= total) return;
    int v = tid & 15;
    int l = (tid >> 4) & 31;
    int rest = tid >> 9;
    int t = rest % ntiles;
    int s = rest / ntiles;
    int n = t*16 + (l & 15);
    int k = s*32 + (v & 7) + ((v >> 3) << 4) + ((l >> 4) << 3);
    frag[tid] = (n < ncols) ? (h16)w[k*ncols + n] : (h16)0.f;
}

// ------------------------------------------------- stage 1: EdgeConv (given edges)
// wave = 16 edges; layer1 (K=6) per-lane VALU -> LDS f16; layer2 = WMMA
// scatter-max via single unsigned atomicMax on flipped floats
__global__ void __launch_bounds__(128) k_conv1(
    const float* __restrict__ x, const long long* __restrict__ ei,
    const float* __restrict__ w1, const float* __restrict__ b1,
    const h16*   __restrict__ w2f, const float* __restrict__ b2,
    unsigned* __restrict__ out)
{
    __shared__ float sw1[6*64], sb1[64], sb2[64];
    __shared__ h16  h1S[4][16*64];
    __shared__ int  dstS[4][16];

    for (int i = threadIdx.x; i < 6*64; i += blockDim.x) sw1[i] = w1[i];
    for (int i = threadIdx.x; i < 64;   i += blockDim.x) { sb1[i] = b1[i]; sb2[i] = b2[i]; }
    __syncthreads();

    int wv = threadIdx.x >> 5, lane = threadIdx.x & 31;
    int tile  = blockIdx.x*4 + wv;
    int eBase = tile*16;
    int r = lane & 15, half = lane >> 4;

    // layer 1: lane covers edge r, outputs [half*32, half*32+32)
    {
        int e = eBase + r;
        int s = (int)ei[e];          // src
        int d = (int)ei[E1 + e];     // dst
        if (lane < 16) dstS[wv][lane] = d;
        float u[6];
        u[0] = x[d*3+0]; u[1] = x[d*3+1]; u[2] = x[d*3+2];
        u[3] = x[s*3+0]-u[0]; u[4] = x[s*3+1]-u[1]; u[5] = x[s*3+2]-u[2];
        #pragma unroll
        for (int oo = 0; oo < 32; ++oo) {
            int o = half*32 + oo;
            float a = sb1[o];
            #pragma unroll
            for (int k = 0; k < 6; ++k) a += u[k] * sw1[k*64+o];
            h1S[wv][r*64 + o] = (h16)(a > 0.f ? a : 0.f);
        }
    }
    // A fragments (16 edges x K=64); same-wave DS ops are in-order
    v16h a2[2];
    #pragma unroll
    for (int s = 0; s < 2; ++s) {
        v16h av;
        #pragma unroll
        for (int v = 0; v < 16; ++v) {
            int kk = (v & 7) + ((v >> 3) << 4) + (half << 3) + s*32;
            av[v] = h1S[wv][r*64 + kk];
        }
        a2[s] = av;
    }
    // layer 2: 4 N-tiles x 2 K-steps + scatter-max epilogue
    #pragma unroll
    for (int t = 0; t < 4; ++t) {
        v8f acc = {};
        #pragma unroll
        for (int s = 0; s < 2; ++s) {
            v16h bf = *reinterpret_cast<const v16h*>(w2f + ((s*4+t)*32 + lane)*16);
            acc = __builtin_amdgcn_wmma_f32_16x16x32_f16(false, a2[s], false, bf, (short)0, acc, false, false);
        }
        int n = t*16 + r;
        float bias = sb2[n];
        #pragma unroll
        for (int v = 0; v < 8; ++v) {
            int mm = v + (half << 3);
            int d  = dstS[wv][mm];
            atomicMax(&out[(size_t)d*64 + n], flipF(acc[v] + bias));
        }
    }
}

// ------------------------------ decode(optional), ELU, f16 convert, ||y||^2
__global__ void k_prep(const float* __restrict__ hin, int encoded,
                       h16* __restrict__ y16, float* __restrict__ sq)
{
    int n = blockIdx.x * blockDim.x + threadIdx.x;
    if (n >= BN_) return;
    const unsigned* uin = (const unsigned*)hin;
    float acc = 0.f;
    for (int f = 0; f < 64; ++f) {
        float v = encoded ? unflipF(uin[n*64+f]) : hin[n*64+f];
        if (v <= -FLT_MAX) v = 0.f;                 // segment_max empty fill -> 0
        float e = v > 0.f ? v : (expf(v) - 1.f);    // ELU(alpha=1)
        acc += e * e;
        y16[n*64+f] = (h16)e;
    }
    sq[n] = acc;
}

// ---------------------------------------- kNN: WMMA gram tiles + register top-K
// block = 8 waves x 16 rows = 128 rows of one graph; loop over 64-column slabs
__global__ void __launch_bounds__(256) k_knn(
    const h16* __restrict__ y16, const float* __restrict__ sq, int* __restrict__ idx)
{
    __shared__ __align__(16) h16 colS[64*64];      // shared 64-col slab [c][f]
    __shared__ float dS[8][16*64];                 // per-wave distance tile
    __shared__ float sqrowS[8][16];

    int wv = threadIdx.x >> 5, lane = threadIdx.x & 31;
    int g  = blockIdx.x >> 4;            // graph id
    int rb = blockIdx.x & 15;            // 128-row block within graph
    int nodeBase = g*NPT + rb*128 + wv*16;

    int m = lane & 15, half = lane >> 4;

    // A fragments (16 rows x K=64): row=m, element v -> k=v+half*16
    v16h a0, a1;
    #pragma unroll
    for (int v = 0; v < 16; ++v) {
        int kk = v + (half << 4);
        a0[v] = y16[(nodeBase+m)*64 + kk];
        a1[v] = y16[(nodeBase+m)*64 + kk + 32];
    }
    if (lane < 16) sqrowS[wv][lane] = sq[nodeBase + lane];

    float kd[KNN]; int ki[KNN];
    #pragma unroll
    for (int t = 0; t < KNN; ++t) { kd[t] = FLT_MAX; ki[t] = 0; }

    for (int ct = 0; ct < NPT/64; ++ct) {
        __syncthreads();                                   // previous slab consumed
        // async-copy the contiguous 8KB slab (64 rows of y16) into LDS
        {
            const h16* slab = y16 + (size_t)(g*NPT + ct*64)*64;
            for (int q = threadIdx.x; q < 512; q += 256)   // 512 x 16B chunks
                asyncCopy16(slab + q*8, colS + q*8);
            // prefetch next slab into cache hierarchy
            if (ct+1 < NPT/64 && threadIdx.x < 64)
                __builtin_prefetch(y16 + (size_t)(g*NPT + (ct+1)*64)*64 + threadIdx.x*64, 0, 1);
            asyncWait0();
        }
        __syncthreads();                                   // slab ready

        #pragma unroll
        for (int c4 = 0; c4 < 4; ++c4) {
            v16h b0, b1;                                   // B: col=m, k=v+half*16
            #pragma unroll
            for (int v = 0; v < 16; ++v) {
                int kk = v + (half << 4);
                b0[v] = colS[(c4*16 + m)*64 + kk];
                b1[v] = colS[(c4*16 + m)*64 + kk + 32];
            }
            v8f acc = {};
            acc = __builtin_amdgcn_wmma_f32_16x16x32_f16(false, a0, false, b0, (short)0, acc, false, false);
            acc = __builtin_amdgcn_wmma_f32_16x16x32_f16(false, a1, false, b1, (short)0, acc, false, false);

            float sqc = sq[g*NPT + ct*64 + c4*16 + m];
            #pragma unroll
            for (int v = 0; v < 8; ++v) {
                int rr = v + (half << 3);
                dS[wv][rr*64 + c4*16 + m] = sqrowS[wv][rr] + sqc - 2.f*acc[v];
            }
        }
        __syncthreads();                                   // dist tiles ready

        if (lane < 16) {                                   // lane owns row `lane`
            #pragma unroll 1
            for (int j = 0; j < 64; ++j) {
                float d = dS[wv][lane*64 + j];
                if (d < kd[KNN-1]) {
                    float vd = d; int vi = g*NPT + ct*64 + j;  // global node id
                    #pragma unroll
                    for (int t = 0; t < KNN; ++t) {
                        if (vd < kd[t]) {
                            float td = kd[t]; int ti = ki[t];
                            kd[t] = vd; ki[t] = vi; vd = td; vi = ti;
                        }
                    }
                }
            }
        }
    }
    if (lane < 16) {
        #pragma unroll
        for (int t = 0; t < KNN; ++t) idx[(nodeBase+lane)*KNN + t] = ki[t];
    }
}

// ------------------- dynamic EdgeConv MLP: wave owns 4 nodes = 80 edges = 5 tiles
// barrier-free; max-reduce in LDS via single ds_max_u32 on flipped floats
__global__ void __launch_bounds__(128) k_dynmlp(
    const h16* __restrict__ y16, const int* __restrict__ idx,
    const h16* __restrict__ w1f, const float* __restrict__ b1,
    const h16* __restrict__ w2f, const float* __restrict__ b2,
    float* __restrict__ out)
{
    __shared__ __align__(16) h16 xiS[4][16*64];   // per-wave xi rows
    __shared__ __align__(16) h16 xjS[4][16*64];   // per-wave xj rows
    __shared__ h16  h1S[4][16*64];                // per-wave hidden after relu
    __shared__ unsigned mS[4][4*64];              // per-wave node max accum (flipped)

    int wv = threadIdx.x >> 5, lane = threadIdx.x & 31;
    int grp = blockIdx.x*4 + wv;                  // node group (4 nodes)
    int q0  = grp*4;                              // first node
    int r = lane & 15, half = lane >> 4;

    #pragma unroll
    for (int i = 0; i < 8; ++i) mS[wv][lane + i*32] = NEG_INF_FLIP;

    for (int sub = 0; sub < 5; ++sub) {
        int eBase = q0*KNN + sub*16;

        // stage xi/xj rows via async global->LDS (lane: edge r, feature half `half`)
        {
            int e  = eBase + r;
            int ni = e / KNN;
            int nj = idx[e];
            const h16* gi = y16 + (size_t)ni*64 + half*32;
            const h16* gj = y16 + (size_t)nj*64 + half*32;
            h16* li = &xiS[wv][r*64 + half*32];
            h16* lj = &xjS[wv][r*64 + half*32];
            #pragma unroll
            for (int c = 0; c < 4; ++c) {
                asyncCopy16(gi + c*8, li + c*8);
                asyncCopy16(gj + c*8, lj + c*8);
            }
            asyncWait0();
        }

        // A fragments for layer 1 (K=128): k<64 -> xi, k>=64 -> xj-xi
        v16h a1[4];
        #pragma unroll
        for (int s = 0; s < 4; ++s) {
            v16h av;
            #pragma unroll
            for (int v = 0; v < 16; ++v) {
                int kk = (v & 7) + ((v >> 3) << 4) + (half << 3) + s*32;
                if (kk < 64) av[v] = xiS[wv][r*64 + kk];
                else         av[v] = (h16)(xjS[wv][r*64 + kk-64] - xiS[wv][r*64 + kk-64]);
            }
            a1[s] = av;
        }
        // layer 1: 4 N-tiles x 4 K-steps
        #pragma unroll
        for (int t = 0; t < 4; ++t) {
            v8f acc = {};
            #pragma unroll
            for (int s = 0; s < 4; ++s) {
                v16h bf = *reinterpret_cast<const v16h*>(w1f + ((s*4+t)*32 + lane)*16);
                acc = __builtin_amdgcn_wmma_f32_16x16x32_f16(false, a1[s], false, bf, (short)0, acc, false, false);
            }
            int n = t*16 + r;
            float bias = b1[n];
            #pragma unroll
            for (int v = 0; v < 8; ++v) {
                int mm = v + (half << 3);
                float hv = acc[v] + bias;
                h1S[wv][mm*64 + n] = (h16)(hv > 0.f ? hv : 0.f);
            }
        }
        // A fragments for layer 2 (K=64)
        v16h a2[2];
        #pragma unroll
        for (int s = 0; s < 2; ++s) {
            v16h av;
            #pragma unroll
            for (int v = 0; v < 16; ++v) {
                int kk = (v & 7) + ((v >> 3) << 4) + (half << 3) + s*32;
                av[v] = h1S[wv][r*64 + kk];
            }
            a2[s] = av;
        }
        // layer 2 + LDS max epilogue (single unsigned ds max per element)
        #pragma unroll
        for (int t = 0; t < 4; ++t) {
            v8f acc = {};
            #pragma unroll
            for (int s = 0; s < 2; ++s) {
                v16h bf = *reinterpret_cast<const v16h*>(w2f + ((s*4+t)*32 + lane)*16);
                acc = __builtin_amdgcn_wmma_f32_16x16x32_f16(false, a2[s], false, bf, (short)0, acc, false, false);
            }
            int n = t*16 + r;
            float bias = b2[n];
            #pragma unroll
            for (int v = 0; v < 8; ++v) {
                int mm = v + (half << 3);
                int e  = eBase + mm;
                int ql = e/KNN - q0;                // local node 0..3
                atomicMax(&mS[wv][ql*64 + n], flipF(acc[v] + bias));
            }
        }
    }
    // write the 4 nodes' features, decoding flip (same-wave DS ops are in-order)
    #pragma unroll
    for (int i = 0; i < 8; ++i) {
        int q = lane + i*32;                       // 0..255 -> node q/64, feat q&63
        out[(size_t)(q0 + (q >> 6))*64 + (q & 63)] = unflipF(mS[wv][q]);
    }
}

// ----------------------------------------------------- final linear 64 -> 40 (WMMA)
// wave = 16 nodes; N padded to 48 (3 n-tiles, zero-filled weight fragments)
__global__ void __launch_bounds__(128) k_final(
    const float* __restrict__ h, const h16* __restrict__ lwf,
    const float* __restrict__ lb, float* __restrict__ out)
{
    int wv = threadIdx.x >> 5, lane = threadIdx.x & 31;
    int tile = blockIdx.x*4 + wv;
    int nodeBase = tile*16;
    int r = lane & 15, half = lane >> 4;

    // A fragments (16 nodes x K=64) from fp32 h, converted to f16
    v16h a0, a1;
    #pragma unroll
    for (int v = 0; v < 16; ++v) {
        int kk = (v & 7) + ((v >> 3) << 4) + (half << 3);
        a0[v] = (h16)h[(size_t)(nodeBase+r)*64 + kk];
        a1[v] = (h16)h[(size_t)(nodeBase+r)*64 + kk + 32];
    }
    #pragma unroll
    for (int t = 0; t < 3; ++t) {
        v8f acc = {};
        {
            v16h bf = *reinterpret_cast<const v16h*>(lwf + ((0*3+t)*32 + lane)*16);
            acc = __builtin_amdgcn_wmma_f32_16x16x32_f16(false, a0, false, bf, (short)0, acc, false, false);
            bf = *reinterpret_cast<const v16h*>(lwf + ((1*3+t)*32 + lane)*16);
            acc = __builtin_amdgcn_wmma_f32_16x16x32_f16(false, a1, false, bf, (short)0, acc, false, false);
        }
        int n = t*16 + r;
        if (n < NCLS) {
            float bias = lb[n];
            #pragma unroll
            for (int v = 0; v < 8; ++v) {
                int mm = v + (half << 3);
                out[(size_t)(nodeBase+mm)*NCLS + n] = acc[v] + bias;
            }
        }
    }
}

// ---------------------------------------------------------------- launcher
extern "C" void kernel_launch(void* const* d_in, const int* in_sizes, int n_in,
                              void* d_out, int out_size, void* d_ws, size_t ws_size,
                              hipStream_t stream)
{
    const float*     x  = (const float*)d_in[0];
    const long long* ei = (const long long*)d_in[1];
    // d_in[2] = batch (unused; graphs are fixed-size)
    const float* c1w1 = (const float*)d_in[3];
    const float* c1b1 = (const float*)d_in[4];
    const float* c1w2 = (const float*)d_in[5];
    const float* c1b2 = (const float*)d_in[6];
    const float* d1w1 = (const float*)d_in[7];
    const float* d1b1 = (const float*)d_in[8];
    const float* d1w2 = (const float*)d_in[9];
    const float* d1b2 = (const float*)d_in[10];
    const float* d2w1 = (const float*)d_in[11];
    const float* d2b1 = (const float*)d_in[12];
    const float* d2w2 = (const float*)d_in[13];
    const float* d2b2 = (const float*)d_in[14];
    const float* lw   = (const float*)d_in[15];
    const float* lb   = (const float*)d_in[16];
    float* out = (float*)d_out;

    // workspace carve-up (256B aligned)
    char* ws = (char*)d_ws; size_t off = 0;
    auto carve = [&](size_t bytes) { void* p = ws + off; off = (off + bytes + 255) & ~(size_t)255; return p; };
    float* hA   = (float*)carve((size_t)BN_*64*sizeof(float));
    float* hB   = (float*)carve((size_t)BN_*64*sizeof(float));
    h16*   y16  = (h16*)  carve((size_t)BN_*64*sizeof(h16));
    float* sq   = (float*)carve((size_t)BN_*sizeof(float));
    int*   idxb = (int*)  carve((size_t)BN_*KNN*sizeof(int));
    h16*   w1f  = (h16*)  carve((size_t)128*64*sizeof(h16));   // dyn layer1 frags
    h16*   w2f  = (h16*)  carve((size_t)64*64*sizeof(h16));    // dyn layer2 frags
    h16*   c2f  = (h16*)  carve((size_t)64*64*sizeof(h16));    // conv1 layer2 frags
    h16*   linf = (h16*)  carve((size_t)64*48*sizeof(h16));    // final linear frags (N pad 48)

    const int nfeat = BN_*64;
    dim3 b256(256), b128(128);

    // ---- stage 1: EdgeConv on given edges -> hA (flipped-uint encoding)
    k_wfrag<<<16, b256, 0, stream>>>(c1w2, c2f, 2, 4, 64);
    k_wfrag<<<12, b256, 0, stream>>>(lw, linf, 2, 3, NCLS);
    k_fillu<<<(nfeat+255)/256, b256, 0, stream>>>((unsigned*)hA, NEG_INF_FLIP, nfeat);
    k_conv1<<<(E1/16)/4, b128, 0, stream>>>(x, ei, c1w1, c1b1, c2f, c1b2, (unsigned*)hA);
    k_prep<<<(BN_+255)/256, b256, 0, stream>>>(hA, 1, y16, sq);

    // ---- stage 2: dynamic EdgeConv #1 -> hB (plain float)
    k_wfrag<<<32, b256, 0, stream>>>(d1w1, w1f, 4, 4, 64);
    k_wfrag<<<16, b256, 0, stream>>>(d1w2, w2f, 2, 4, 64);
    k_knn<<<BGR*(NPT/128), b256, 0, stream>>>(y16, sq, idxb);
    k_dynmlp<<<(BN_/4)/4, b128, 0, stream>>>(y16, idxb, w1f, d1b1, w2f, d1b2, hB);
    k_prep<<<(BN_+255)/256, b256, 0, stream>>>(hB, 0, y16, sq);

    // ---- stage 3: dynamic EdgeConv #2 -> hA (plain float)
    k_wfrag<<<32, b256, 0, stream>>>(d2w1, w1f, 4, 4, 64);
    k_wfrag<<<16, b256, 0, stream>>>(d2w2, w2f, 2, 4, 64);
    k_knn<<<BGR*(NPT/128), b256, 0, stream>>>(y16, sq, idxb);
    k_dynmlp<<<(BN_/4)/4, b128, 0, stream>>>(y16, idxb, w1f, d2b1, w2f, d2b2, hA);

    // ---- final linear (WMMA) -> d_out
    k_final<<<(BN_/16)/4, b128, 0, stream>>>(hA, linf, lb, out);
}
